// SSVA_Net_35966056137231
// MI455X (gfx1250) — compile-verified, matrix operand order checked by reading, not run on hardware
//
#include <hip/hip_runtime.h>
#include <math.h>

typedef __attribute__((ext_vector_type(16))) _Float16 v16h;
typedef __attribute__((ext_vector_type(8)))  _Float16 v8h;
typedef __attribute__((ext_vector_type(8)))  float    v8f;

#define TOKENS 131072
#define CDIM   180
#define CP     192     // padded K for C=180 (6 chunks of 32)
#define HDIM   720
#define HP     736     // padded K for 720 (23 chunks of 32)
#define NHEADS 6
#define DHEAD  30
#define IMG    256
#define WIN    8
#define QSTR   576     // qkv row stride   (540 -> 9 n-tiles of 64)
#define HSTR   768     // hidden row stride(720 -> 12 n-tiles of 64)

// ---- swizzle weights f32[K,N] -> f16 WMMA-fragment order [chunk][ntile][lane][16] ----
__global__ void swizzle_w_kernel(const float* __restrict__ src, _Float16* __restrict__ dst,
                                 int K, int N, int nTiles, int total) {
  int idx = blockIdx.x * blockDim.x + threadIdx.x;
  if (idx >= total) return;
  int j     = idx & 15;
  int lane  = (idx >> 4) & 31;
  int tile  = (idx >> 9) % nTiles;
  int chunk = idx / (nTiles << 9);
  int n = tile * 16 + (lane & 15);
  int k = chunk * 32 + ((lane >> 4) << 3) + j + ((j >= 8) ? 8 : 0);
  float v = (k < K && n < N) ? src[(size_t)k * N + n] : 0.f;
  dst[idx] = (_Float16)v;
}

// ---- precompute rel-pos bias in WMMA C-layout: [h][mt][nt][lane][8] (f32) ----
__global__ void swizzle_bias_kernel(const float* __restrict__ rpb, float* __restrict__ dst,
                                    int total) {
  int idx = blockIdx.x * blockDim.x + threadIdx.x;
  if (idx >= total) return;
  int v    = idx & 7;
  int lane = (idx >> 3) & 31;
  int nt   = (idx >> 8) & 3;
  int mt   = (idx >> 10) & 3;
  int h    = idx >> 12;
  int m = mt * 16 + ((lane >> 4) << 3) + v;   // query index in window
  int n = nt * 16 + (lane & 15);              // key index in window
  int dy = (m >> 3) - (n >> 3) + (WIN - 1);
  int dx = (m & 7)  - (n & 7)  + (WIN - 1);
  dst[idx] = rpb[(dy * (2 * WIN - 1) + dx) * NHEADS + h];
}

// ---- LayerNorm over C, one wave per token, zero-fills pad cols up to Cp ----
__global__ void layernorm_f16_kernel(const float* __restrict__ x,
                                     const float* __restrict__ g,
                                     const float* __restrict__ b,
                                     _Float16* __restrict__ y, int C, int Cp) {
  const int tok = blockIdx.x;
  const float* xr = x + (size_t)tok * C;
  float s = 0.f, ss = 0.f;
  for (int c = threadIdx.x; c < C; c += 32) { float v = xr[c]; s += v; ss += v * v; }
  for (int o = 16; o > 0; o >>= 1) { s += __shfl_xor(s, o, 32); ss += __shfl_xor(ss, o, 32); }
  const float mu   = s / (float)C;
  const float var  = ss / (float)C - mu * mu;
  const float rstd = rsqrtf(var + 1e-5f);
  _Float16* yr = y + (size_t)tok * Cp;
  for (int c = threadIdx.x; c < Cp; c += 32)
    yr[c] = (c < C) ? (_Float16)((xr[c] - mu) * rstd * g[c] + b[c]) : (_Float16)0.f;
}

// ---- WMMA GEMM, no LDS (A from padded f16 activations, B pre-swizzled) ----
__global__ void __launch_bounds__(256)
gemm_wmma_kernel(const _Float16* __restrict__ A, const _Float16* __restrict__ Bsw,
                 const float* __restrict__ bias, _Float16* __restrict__ outH,
                 float* __restrict__ outAdd,
                 int nK, int lda, int N, int Nfill, int ldo, int chunkStride,
                 int mode, int gelu) {
  const int lane  = threadIdx.x & 31;
  const int wave  = threadIdx.x >> 5;
  const int idx15 = lane & 15;
  const int khalf = (lane >> 4) << 3;
  const int m0 = blockIdx.x * 128 + wave * 16;
  const int n0 = blockIdx.y * 64;

  const _Float16* ap = A + (size_t)(m0 + idx15) * lda + khalf;
  const _Float16* bp = Bsw + (size_t)blockIdx.y * 2048 + (size_t)lane * 16;

  v8f acc[4] = {};
  union AF { v16h v; v8h h[2]; };
  for (int kk = 0; kk < nK; ++kk) {
    AF a;
    a.h[0] = *(const v8h*)(ap);
    a.h[1] = *(const v8h*)(ap + 16);
    const v16h b0 = *(const v16h*)(bp);
    const v16h b1 = *(const v16h*)(bp + 512);
    const v16h b2 = *(const v16h*)(bp + 1024);
    const v16h b3 = *(const v16h*)(bp + 1536);
    acc[0] = __builtin_amdgcn_wmma_f32_16x16x32_f16(false, a.v, false, b0, (short)0, acc[0], false, false);
    acc[1] = __builtin_amdgcn_wmma_f32_16x16x32_f16(false, a.v, false, b1, (short)0, acc[1], false, false);
    acc[2] = __builtin_amdgcn_wmma_f32_16x16x32_f16(false, a.v, false, b2, (short)0, acc[2], false, false);
    acc[3] = __builtin_amdgcn_wmma_f32_16x16x32_f16(false, a.v, false, b3, (short)0, acc[3], false, false);
    ap += 32;
    bp += chunkStride;
  }
  const int mrow = m0 + ((lane >> 4) << 3);
  #pragma unroll
  for (int nt = 0; nt < 4; ++nt) {
    const int gn = n0 + nt * 16 + idx15;
    if (gn < Nfill) {
      const float bv = (gn < N) ? bias[gn] : 0.f;
      #pragma unroll
      for (int v = 0; v < 8; ++v) {
        const int gm = mrow + v;
        float r = acc[nt][v] + bv;
        if (mode == 0) {
          if (gelu) r = 0.5f * r * (1.0f + erff(r * 0.70710678118654752f));
          outH[(size_t)gm * ldo + gn] = (_Float16)((gn < N) ? r : 0.f);
        } else if (gn < N) {
          outAdd[(size_t)gm * ldo + gn] += r;
        }
      }
    }
  }
}

// ---- WMMA windowed attention: 1 wave per head, 2 heads per 64-thread block ----
__device__ __forceinline__ int tok_of(int bb, int wy, int wx, int idx, int shift) {
  int r = idx >> 3, c = idx & 7;
  int hh = (wy * WIN + r + shift) & (IMG - 1);
  int ww = (wx * WIN + c + shift) & (IMG - 1);
  return bb * (IMG * IMG) + hh * IMG + ww;
}

__device__ __forceinline__ v16h load_qk_frag(const _Float16* p, int khalf) {
  union AF { v16h v; v8h h[2]; } a;
  a.h[0] = *(const v8h*)(p);
  a.h[1] = *(const v8h*)(p + 16);
  if (khalf) { a.h[1][6] = (_Float16)0.f; a.h[1][7] = (_Float16)0.f; }  // k=30,31 pad
  return a.v;
}

#define SPS 72  // LDS row stride (halfs): 144B, keeps b128 fragment reads 16B-aligned

__global__ void __launch_bounds__(64)
win_attn_wmma_kernel(const _Float16* __restrict__ qkv, const float* __restrict__ bias_sw,
                     _Float16* __restrict__ out, int shift) {
  __shared__ __align__(16) _Float16 Sp[2][64][SPS];  // P matrix bounce (per wave)
  __shared__ __align__(16) _Float16 Vt[2][32][SPS];  // V transposed [feat][key]

  const int wi   = blockIdx.x;             // 0..2047 windows
  const int wv   = threadIdx.x >> 5;       // wave in block: 0/1
  const int h    = blockIdx.y * 2 + wv;    // head 0..5
  const int lane = threadIdx.x & 31;
  const int idx15 = lane & 15;
  const int g     = lane >> 4;
  const int khalf = g << 3;
  const int bb  = wi >> 10;
  const int rem = wi & 1023;
  const int wy  = rem >> 5;
  const int wx  = rem & 31;

  // ---- stage V transposed into LDS (2 keys per lane), pad feats 30,31 with 0 ----
  #pragma unroll
  for (int kk = 0; kk < 2; ++kk) {
    const int key = lane + kk * 32;
    const _Float16* vr = qkv + (size_t)tok_of(bb, wy, wx, key, shift) * QSTR + 2 * CDIM + h * DHEAD;
    #pragma unroll
    for (int f = 0; f < DHEAD; ++f) Vt[wv][f][key] = vr[f];
    Vt[wv][30][key] = (_Float16)0.f;
    Vt[wv][31][key] = (_Float16)0.f;
  }

  // ---- S = Q K^T : 16 wmma tiles, single K-chunk (d=30 padded to 32) ----
  v16h qf[4];
  #pragma unroll
  for (int mt = 0; mt < 4; ++mt) {
    const int tq = tok_of(bb, wy, wx, mt * 16 + idx15, shift);
    qf[mt] = load_qk_frag(qkv + (size_t)tq * QSTR + h * DHEAD + khalf, khalf);
  }
  v8f sacc[4][4] = {};
  #pragma unroll
  for (int nt = 0; nt < 4; ++nt) {
    const int tk = tok_of(bb, wy, wx, nt * 16 + idx15, shift);
    const v16h kf = load_qk_frag(qkv + (size_t)tk * QSTR + CDIM + h * DHEAD + khalf, khalf);
    #pragma unroll
    for (int mt = 0; mt < 4; ++mt)
      sacc[mt][nt] = __builtin_amdgcn_wmma_f32_16x16x32_f16(false, qf[mt], false, kf,
                                                            (short)0, sacc[mt][nt], false, false);
  }

  // ---- scale + swizzled bias ----
  const float sc = 0.18257418583505536f;  // 30^-0.5
  #pragma unroll
  for (int mt = 0; mt < 4; ++mt)
    #pragma unroll
    for (int nt = 0; nt < 4; ++nt) {
      const v8f bv = *(const v8f*)(bias_sw + (((h * 4 + mt) * 4 + nt) * 32 + lane) * 8);
      #pragma unroll
      for (int e = 0; e < 8; ++e) sacc[mt][nt][e] = sacc[mt][nt][e] * sc + bv[e];
    }

  // ---- softmax per row (row = fixed g,v; reduce 4 n-tiles x 16 lanes of same half) ----
  #pragma unroll
  for (int mt = 0; mt < 4; ++mt) {
    #pragma unroll
    for (int v = 0; v < 8; ++v) {
      float rm = fmaxf(fmaxf(sacc[mt][0][v], sacc[mt][1][v]),
                       fmaxf(sacc[mt][2][v], sacc[mt][3][v]));
      rm = fmaxf(rm, __shfl_xor(rm, 1, 32));
      rm = fmaxf(rm, __shfl_xor(rm, 2, 32));
      rm = fmaxf(rm, __shfl_xor(rm, 4, 32));
      rm = fmaxf(rm, __shfl_xor(rm, 8, 32));
      float e0 = __expf(sacc[mt][0][v] - rm);
      float e1 = __expf(sacc[mt][1][v] - rm);
      float e2 = __expf(sacc[mt][2][v] - rm);
      float e3 = __expf(sacc[mt][3][v] - rm);
      float rs = e0 + e1 + e2 + e3;
      rs += __shfl_xor(rs, 1, 32);
      rs += __shfl_xor(rs, 2, 32);
      rs += __shfl_xor(rs, 4, 32);
      rs += __shfl_xor(rs, 8, 32);
      const float inv = 1.0f / rs;
      const int m = mt * 16 + g * 8 + v;
      Sp[wv][m][ 0 + idx15] = (_Float16)(e0 * inv);
      Sp[wv][m][16 + idx15] = (_Float16)(e1 * inv);
      Sp[wv][m][32 + idx15] = (_Float16)(e2 * inv);
      Sp[wv][m][48 + idx15] = (_Float16)(e3 * inv);
    }
  }
  __syncthreads();

  // ---- O = P V : K = 64 keys (2 chunks), N = 32 feats (2 tiles) ----
  v8f oacc[4][2] = {};
  #pragma unroll
  for (int ck = 0; ck < 2; ++ck) {
    const int kb = ck * 32 + khalf;
    union AF { v16h v; v8h h[2]; } bfr[2];
    #pragma unroll
    for (int ntv = 0; ntv < 2; ++ntv) {
      const _Float16* bp = &Vt[wv][ntv * 16 + idx15][kb];
      bfr[ntv].h[0] = *(const v8h*)(bp);
      bfr[ntv].h[1] = *(const v8h*)(bp + 16);
    }
    #pragma unroll
    for (int mt = 0; mt < 4; ++mt) {
      union AF { v16h v; v8h h[2]; } a;
      const _Float16* ap = &Sp[wv][mt * 16 + idx15][kb];
      a.h[0] = *(const v8h*)(ap);
      a.h[1] = *(const v8h*)(ap + 16);
      #pragma unroll
      for (int ntv = 0; ntv < 2; ++ntv)
        oacc[mt][ntv] = __builtin_amdgcn_wmma_f32_16x16x32_f16(false, a.v, false, bfr[ntv].v,
                                                               (short)0, oacc[mt][ntv], false, false);
    }
  }

  // ---- scatter output (feat < 30) ----
  #pragma unroll
  for (int mt = 0; mt < 4; ++mt) {
    #pragma unroll
    for (int ntv = 0; ntv < 2; ++ntv) {
      const int feat = ntv * 16 + idx15;
      if (feat < DHEAD) {
        #pragma unroll
        for (int v = 0; v < 8; ++v) {
          const int m = mt * 16 + g * 8 + v;
          const size_t to = (size_t)tok_of(bb, wy, wx, m, shift) * CP + h * DHEAD + feat;
          out[to] = (_Float16)oacc[mt][ntv][v];
        }
      }
    }
  }
}

// ---------------- host orchestration ----------------
extern "C" void kernel_launch(void* const* d_in, const int* in_sizes, int n_in,
                              void* d_out, int out_size, void* d_ws, size_t ws_size,
                              hipStream_t stream) {
  const float* x_in   = (const float*)d_in[0];
  const float* ln1_g  = (const float*)d_in[1];
  const float* ln1_b  = (const float*)d_in[2];
  const float* qkv_w  = (const float*)d_in[3];
  const float* qkv_b  = (const float*)d_in[4];
  const float* rpb    = (const float*)d_in[5];
  const float* proj_w = (const float*)d_in[6];
  const float* proj_b = (const float*)d_in[7];
  const float* ln2_g  = (const float*)d_in[8];
  const float* ln2_b  = (const float*)d_in[9];
  const float* fc1_w  = (const float*)d_in[10];
  const float* fc1_b  = (const float*)d_in[11];
  const float* fc2_w  = (const float*)d_in[12];
  const float* fc2_b  = (const float*)d_in[13];
  float* x = (float*)d_out;  // running residual

  // ws layout (bytes):
  //   [0 .. 1,744,896)      swizzled f16 weights
  //   [1,744,896 .. +196,608) swizzled bias f32 (2 blocks x 24576 floats)
  //   [2MB .. +50,331,648)  yln / attn-out (f16, stride 192)
  //   [52,428,800 .. )      qkv / MLP-hidden (f16, stride 576/768)
  _Float16* w16     = (_Float16*)d_ws;
  _Float16* qkv_sw  = w16;                  // 2*110592
  _Float16* proj_sw = w16 + 221184;         // 2*36864
  _Float16* fc1_sw  = w16 + 294912;         // 2*147456
  _Float16* fc2_sw  = w16 + 589824;         // 2*141312 -> end 872448 halfs
  float*    bias_sw = (float*)((char*)d_ws + 1744896);
  _Float16* yln    = (_Float16*)((char*)d_ws + (size_t)(2 << 20));
  _Float16* qkvbuf = (_Float16*)((char*)d_ws + (size_t)(2 << 20)
                                 + (size_t)TOKENS * CP * sizeof(_Float16));

  for (int i = 0; i < 2; ++i) {
    int t;
    t = 110592;
    swizzle_w_kernel<<<(t + 255) / 256, 256, 0, stream>>>(
        qkv_w + (size_t)i * CDIM * 3 * CDIM, qkv_sw + (size_t)i * t, CDIM, 3 * CDIM, 36, t);
    t = 36864;
    swizzle_w_kernel<<<(t + 255) / 256, 256, 0, stream>>>(
        proj_w + (size_t)i * CDIM * CDIM, proj_sw + (size_t)i * t, CDIM, CDIM, 12, t);
    t = 147456;
    swizzle_w_kernel<<<(t + 255) / 256, 256, 0, stream>>>(
        fc1_w + (size_t)i * CDIM * HDIM, fc1_sw + (size_t)i * t, CDIM, HDIM, 48, t);
    t = 141312;
    swizzle_w_kernel<<<(t + 255) / 256, 256, 0, stream>>>(
        fc2_w + (size_t)i * HDIM * CDIM, fc2_sw + (size_t)i * t, HDIM, CDIM, 12, t);
    t = NHEADS * 4 * 4 * 32 * 8;  // 24576
    swizzle_bias_kernel<<<(t + 255) / 256, 256, 0, stream>>>(
        rpb + (size_t)i * (2 * WIN - 1) * (2 * WIN - 1) * NHEADS, bias_sw + (size_t)i * t, t);
  }

  hipMemcpyAsync(x, x_in, (size_t)TOKENS * CDIM * sizeof(float),
                 hipMemcpyDeviceToDevice, stream);

  for (int i = 0; i < 2; ++i) {
    const int shift = i ? (WIN / 2) : 0;

    layernorm_f16_kernel<<<TOKENS, 32, 0, stream>>>(
        x, ln1_g + i * CDIM, ln1_b + i * CDIM, yln, CDIM, CP);

    gemm_wmma_kernel<<<dim3(TOKENS / 128, 9), 256, 0, stream>>>(
        yln, qkv_sw + (size_t)i * 110592, qkv_b + i * 3 * CDIM,
        qkvbuf, nullptr, 6, CP, 540, 540, QSTR, QSTR * 32, 0, 0);

    win_attn_wmma_kernel<<<dim3(2048, 3), 64, 0, stream>>>(
        qkvbuf, bias_sw + (size_t)i * 24576, yln, shift);

    gemm_wmma_kernel<<<dim3(TOKENS / 128, 3), 256, 0, stream>>>(
        yln, proj_sw + (size_t)i * 36864, proj_b + i * CDIM,
        nullptr, x, 6, CP, CDIM, CDIM, CDIM, CP * 32, 1, 0);

    layernorm_f16_kernel<<<TOKENS, 32, 0, stream>>>(
        x, ln2_g + i * CDIM, ln2_b + i * CDIM, yln, CDIM, CP);

    const int CH = TOKENS / 2;
    for (int ch = 0; ch < 2; ++ch) {
      const size_t to = (size_t)ch * CH;
      gemm_wmma_kernel<<<dim3(CH / 128, 12), 256, 0, stream>>>(
          yln + to * CP, fc1_sw + (size_t)i * 147456, fc1_b + i * HDIM,
          qkvbuf, nullptr, 6, CP, HDIM, HP, HSTR, HSTR * 32, 0, 1);
      gemm_wmma_kernel<<<dim3(CH / 128, 3), 256, 0, stream>>>(
          qkvbuf, fc2_sw + (size_t)i * 141312, fc2_b + i * CDIM,
          nullptr, x + to * CDIM, 23, HSTR, CDIM, CDIM, CDIM, CP * 32, 1, 0);
    }
  }
}
